// InternLM2VEDecoderLayer_6459630814151
// MI455X (gfx1250) — matrix operand order repaired and checked by reading, not compiled, not run on hardware
//
#include <hip/hip_runtime.h>

// ---------------------------------------------------------------------------
// Types for CDNA5 WMMA (wave32, v_wmma_f32_16x16x32_bf16)
// ---------------------------------------------------------------------------
typedef __bf16 bf16_t;
typedef __attribute__((ext_vector_type(16))) __bf16 v16bf;
typedef __attribute__((ext_vector_type(8)))  __bf16 v8bf;
typedef __attribute__((ext_vector_type(8)))  float  v8f;

#define TT 2048
#define DD 2048
#define HH 16
#define KK 8
#define HD 128
#define II 8192

__device__ __forceinline__ v16bf load_frag(const bf16_t* p0, const bf16_t* p1) {
  union { v16bf v; v8bf h[2]; } u;
  u.h[0] = *(const v8bf*)p0;
  u.h[1] = *(const v8bf*)p1;
  return u.v;
}

__device__ __forceinline__ v8f wmma_bf16(v16bf a, v16bf b, v8f c) {
  return __builtin_amdgcn_wmma_f32_16x16x32_bf16(false, a, false, b, (short)0, c,
                                                 false, false);
}

// CDNA5 async global -> LDS copy (16 bytes per lane), tracked by ASYNCcnt.
__device__ __forceinline__ void async_ld16(const bf16_t* g, const bf16_t* l) {
  const unsigned loff = (unsigned)(size_t)l;  // generic LDS addr[31:0] = LDS offset
  asm volatile("global_load_async_to_lds_b128 %0, %1, off"
               :: "v"(loff), "v"(g) : "memory");
}

template <int N>
__device__ __forceinline__ void wait_async() {
  asm volatile("s_wait_asynccnt %0" :: "i"(N) : "memory");
}

__device__ __forceinline__ void wait_ds0() {
  asm volatile("s_wait_dscnt 0x0" ::: "memory");
}

// ---------------------------------------------------------------------------
// DPP16 butterfly reductions within 16-lane halves (no LDS, no dscnt stalls).
// 0xB1 = quad_perm(1,0,3,2), 0x4E = quad_perm(2,3,0,1),
// 0x141 = row_half_mirror,   0x140 = row_mirror.
// ---------------------------------------------------------------------------
template <int CTRL>
__device__ __forceinline__ float dpp_movf(float x) {
  return __int_as_float(__builtin_amdgcn_update_dpp(
      0, __float_as_int(x), CTRL, 0xf, 0xf, true));
}

__device__ __forceinline__ float red_max16(float x) {
  x = fmaxf(x, dpp_movf<0xB1>(x));
  x = fmaxf(x, dpp_movf<0x4E>(x));
  x = fmaxf(x, dpp_movf<0x141>(x));
  x = fmaxf(x, dpp_movf<0x140>(x));
  return x;
}

__device__ __forceinline__ float red_sum16(float x) {
  x += dpp_movf<0xB1>(x);
  x += dpp_movf<0x4E>(x);
  x += dpp_movf<0x141>(x);
  x += dpp_movf<0x140>(x);
  return x;
}

// ---------------------------------------------------------------------------
// Weight convert + transpose: Wt[n][k] = (bf16) W[k][n].  32x32 LDS tiles.
// ---------------------------------------------------------------------------
__global__ __launch_bounds__(256)
void convT_kernel(const float* __restrict__ W, bf16_t* __restrict__ Wt,
                  int Kd, int N) {
  __shared__ float tile[32][33];
  const int n0 = blockIdx.x * 32;
  const int k0 = blockIdx.y * 32;
  const int lt = threadIdx.x & 31;
  const int lr = threadIdx.x >> 5;  // 0..7
#pragma unroll
  for (int i = 0; i < 4; ++i) {
    const int r = lr * 4 + i;
    tile[r][lt] = W[(size_t)(k0 + r) * N + n0 + lt];
  }
  __syncthreads();
#pragma unroll
  for (int i = 0; i < 4; ++i) {
    const int r = lr * 4 + i;
    Wt[(size_t)(n0 + r) * Kd + k0 + lt] = (bf16_t)tile[lt][r];
  }
}

// ---------------------------------------------------------------------------
// Fused residual-add + RMSNorm. r_out (fp32) = x + res ; h_out (bf16) = norm.
// ---------------------------------------------------------------------------
__global__ __launch_bounds__(256)
void addnorm_kernel(const float* __restrict__ x, const float* __restrict__ res,
                    const float* __restrict__ w, float* __restrict__ r_out,
                    bf16_t* __restrict__ h_out) {
  const int t = blockIdx.x;
  const int tid = threadIdx.x;
  __shared__ float red[256];
  const float* xr = x + (size_t)t * DD;
  const float* rr = res + (size_t)t * DD;
  float vals[8];
  float ss = 0.f;
#pragma unroll
  for (int i = 0; i < 8; ++i) {
    const int c = tid + i * 256;
    const float v = xr[c] + rr[c];
    vals[i] = v;
    ss += v * v;
  }
  red[tid] = ss;
  __syncthreads();
  for (int s = 128; s > 0; s >>= 1) {
    if (tid < s) red[tid] += red[tid + s];
    __syncthreads();
  }
  const float rs = rsqrtf(red[0] / (float)DD + 1e-6f);
  float* ro = r_out + (size_t)t * DD;
  bf16_t* ho = h_out + (size_t)t * DD;
#pragma unroll
  for (int i = 0; i < 8; ++i) {
    const int c = tid + i * 256;
    ro[c] = vals[i];
    ho[c] = (bf16_t)(vals[i] * rs * w[c]);
  }
}

// ---------------------------------------------------------------------------
// GEMM: C[M,N] (+)= A[M,Kd] @ Bt[N,Kd]^T, bf16 operands, fp32 accumulate.
// Block: 256 threads (8 waves), tile 128x128, BK=32, double-buffered async LDS.
// Wave: 32x64 (2x4 fragment grid).
// ---------------------------------------------------------------------------
#define BM 128
#define BN 128
#define BK 32
#define LDT 40

template <bool ACCUM>
__global__ __launch_bounds__(256)
void gemm_bf16(const bf16_t* __restrict__ A, const bf16_t* __restrict__ Bt,
               float* __restrict__ C, int M, int N, int Kd) {
  __shared__ alignas(16) bf16_t As[2][BM][LDT];
  __shared__ alignas(16) bf16_t Bs[2][BN][LDT];

  const int tid = threadIdx.x;
  const int lane = tid & 31;
  const int wave = tid >> 5;
  const int bm = blockIdx.y * BM;
  const int bn = blockIdx.x * BN;
  const int wm = (wave >> 1) * 32;  // 0,32,64,96
  const int wn = (wave & 1) * 64;   // 0,64

  const int sr = tid >> 2;        // 0..63 staging row base
  const int sc = (tid & 3) * 8;   // 16B chunk within 32-elem row

  auto issue = [&](int buf, int k0) {
#pragma unroll
    for (int j = 0; j < 2; ++j) {
      const int r = sr + j * 64;
      async_ld16(A + (size_t)(bm + r) * Kd + k0 + sc, &As[buf][r][sc]);
    }
#pragma unroll
    for (int j = 0; j < 2; ++j) {
      const int r = sr + j * 64;
      async_ld16(Bt + (size_t)(bn + r) * Kd + k0 + sc, &Bs[buf][r][sc]);
    }
  };

  const v8f vzero = {0.f, 0.f, 0.f, 0.f, 0.f, 0.f, 0.f, 0.f};
  v8f acc[2][4];
#pragma unroll
  for (int i = 0; i < 2; ++i)
#pragma unroll
    for (int j = 0; j < 4; ++j) acc[i][j] = vzero;

  const int r0 = lane & 15;
  const int klo = (lane >> 4) * 8;
  const int kb16 = (lane >> 4) * 16;

  const int nk = Kd / BK;
  issue(0, 0);
  for (int ik = 0; ik < nk; ++ik) {
    const int buf = ik & 1;
    if (ik + 1 < nk) {
      issue(buf ^ 1, (ik + 1) * BK);
      wait_async<4>();   // in-order: <=4 outstanding => current buffer landed
    } else {
      wait_async<0>();
    }
    __syncthreads();

    v16bf af[2], bfr[4];
#pragma unroll
    for (int i = 0; i < 2; ++i)
      af[i] = load_frag(&As[buf][wm + i * 16 + r0][klo],
                        &As[buf][wm + i * 16 + r0][16 + klo]);
#pragma unroll
    for (int j = 0; j < 4; ++j)
      bfr[j] = load_frag(&Bs[buf][wn + j * 16 + r0][klo],
                         &Bs[buf][wn + j * 16 + r0][16 + klo]);
#pragma unroll
    for (int i = 0; i < 2; ++i)
#pragma unroll
      for (int j = 0; j < 4; ++j) acc[i][j] = wmma_bf16(af[i], bfr[j], acc[i][j]);

    __syncthreads();  // protect buffer reuse by next iteration's issue
  }

  const int half = (lane >> 4) * 8;
  const int cl = lane & 15;
#pragma unroll
  for (int i = 0; i < 2; ++i)
#pragma unroll
    for (int j = 0; j < 4; ++j)
#pragma unroll
      for (int v = 0; v < 8; ++v) {
        const size_t off =
            (size_t)(bm + wm + i * 16 + half + v) * N + (bn + wn + j * 16 + cl);
        float r = acc[i][j][v];
        if (ACCUM) r += C[off];
        C[off] = r;
      }
}

// ---------------------------------------------------------------------------
// Neox RoPE in-place on Q (heads 0..15) and K (heads 16..23) of fp32 qkv.
// ---------------------------------------------------------------------------
__global__ __launch_bounds__(256)
void rope_kernel(float* __restrict__ qkv, const int* __restrict__ pos) {
  const int t = blockIdx.x;
  const float p = (float)pos[t];
  float* row = qkv + (size_t)t * 4096;
  const float lnth = 9.210340371976184f;  // ln(10000)
  for (int i = threadIdx.x; i < (HH + KK) * 64; i += 256) {
    const int head = i >> 6;
    const int j = i & 63;
    const float inv = __expf(-(float)j * (lnth / 64.f));
    const float f = p * inv;
    const float c = __cosf(f);
    const float s = __sinf(f);
    float* b = row + head * HD;
    const float x1 = b[j];
    const float x2 = b[j + 64];
    b[j] = x1 * c - x2 * s;
    b[j + 64] = x2 * c + x1 * s;
  }
}

// ---------------------------------------------------------------------------
// Pack post-RoPE K/V to bf16: kp[kh][t][d], vtp[kh][d][t] (V transposed).
// ---------------------------------------------------------------------------
__global__ __launch_bounds__(256)
void packkv_kernel(const float* __restrict__ qkv, bf16_t* __restrict__ kp,
                   bf16_t* __restrict__ vtp) {
  const int t = blockIdx.x;
  for (int i = threadIdx.x; i < KK * HD; i += 256) {
    const int kh = i >> 7, d = i & 127;
    kp[((size_t)kh * TT + t) * HD + d] = (bf16_t)qkv[(size_t)t * 4096 + 2048 + i];
    vtp[((size_t)kh * HD + d) * TT + t] = (bf16_t)qkv[(size_t)t * 4096 + 3072 + i];
  }
}

// ---------------------------------------------------------------------------
// Causal GQA flash attention. 4 waves/block, wave = 16 query rows x HD=128.
// K/V tiles async-staged double-buffered from packed bf16 buffers.
// ---------------------------------------------------------------------------
#define LQ 136
#define LK 136
#define LV 40
#define LP 40

__global__ __launch_bounds__(128)
void attn_kernel(const float* __restrict__ qkv, const bf16_t* __restrict__ kp,
                 const bf16_t* __restrict__ vtp, bf16_t* __restrict__ out) {
  __shared__ alignas(16) bf16_t Qs[4][16][LQ];
  __shared__ alignas(16) bf16_t Ks[2][32][LK];
  __shared__ alignas(16) bf16_t Vt[2][HD][LV];
  __shared__ alignas(16) bf16_t Ps[4][16][LP];

  const int tid = threadIdx.x;
  const int lane = tid & 31;
  const int w = tid >> 5;
  const int h = blockIdx.y;
  const int kh = h >> 1;  // H/K = 2
  const int q0 = (blockIdx.x * 4 + w) * 16;
  const float scale = 0.08838834764831845f;  // 1/sqrt(128)

  auto issue_kv = [&](int buf, int kb) {
#pragma unroll
    for (int j = 0; j < 4; ++j) {
      const int id = tid + j * 128;
      const int r = id >> 4, c = (id & 15) * 8;
      async_ld16(kp + ((size_t)kh * TT + kb + r) * HD + c, &Ks[buf][r][c]);
    }
#pragma unroll
    for (int j = 0; j < 4; ++j) {
      const int id = tid + j * 128;
      const int d = id >> 2, c = (id & 3) * 8;
      async_ld16(vtp + ((size_t)kh * HD + d) * TT + kb + c, &Vt[buf][d][c]);
    }
  };

  // stage this wave's Q tile (scaled, bf16)
  for (int i = lane; i < 16 * HD; i += 32) {
    const int r = i >> 7, c = i & 127;
    Qs[w][r][c] = (bf16_t)(qkv[(size_t)(q0 + r) * 4096 + h * HD + c] * scale);
  }
  wait_ds0();

  const int r0 = lane & 15;
  const int klo = (lane >> 4) * 8;
  const int kb16 = (lane >> 4) * 16;
  const int rowoff = (lane >> 4) * 8;

  v16bf qa[4];
#pragma unroll
  for (int d = 0; d < 4; ++d)
    qa[d] = load_frag(&Qs[w][r0][d * 32 + klo], &Qs[w][r0][d * 32 + 16 + klo]);

  const v8f vzero = {0.f, 0.f, 0.f, 0.f, 0.f, 0.f, 0.f, 0.f};
  v8f o[8];
  float m[8], l[8], alpha[8];
#pragma unroll
  for (int v = 0; v < 8; ++v) { m[v] = -1e30f; l[v] = 0.f; }
#pragma unroll
  for (int c = 0; c < 8; ++c) o[c] = vzero;

  const int nkb = ((blockIdx.x + 1) * 64) / 32;  // key blocks needed (causal)
  issue_kv(0, 0);
  for (int ib = 0; ib < nkb; ++ib) {
    const int buf = ib & 1;
    const int kb = ib * 32;
    if (ib + 1 < nkb) {
      issue_kv(buf ^ 1, kb + 32);
      wait_async<8>();
    } else {
      wait_async<0>();
    }
    __syncthreads();

    // S = Q K^T (pre-scaled): preload all 8 K fragments, then 8 back-to-back WMMAs
    v16bf kf[2][4];
#pragma unroll
    for (int ct = 0; ct < 2; ++ct)
#pragma unroll
      for (int d = 0; d < 4; ++d)
        kf[ct][d] = load_frag(&Ks[buf][ct * 16 + r0][d * 32 + klo],
                              &Ks[buf][ct * 16 + r0][d * 32 + 16 + klo]);
    v8f s[2];
    s[0] = vzero; s[1] = vzero;
#pragma unroll
    for (int ct = 0; ct < 2; ++ct)
#pragma unroll
      for (int d = 0; d < 4; ++d) s[ct] = wmma_bf16(qa[d], kf[ct][d], s[ct]);

    // causal mask + online softmax (DPP16 butterfly reductions)
#pragma unroll
    for (int v = 0; v < 8; ++v) {
      const int irow = q0 + rowoff + v;
      const int j0 = kb + (lane & 15);
      if (j0 > irow) s[0][v] = -1e30f;
      if (j0 + 16 > irow) s[1][v] = -1e30f;
      const float sv = red_max16(fmaxf(s[0][v], s[1][v]));
      const float nm = fmaxf(m[v], sv);
      const float a_ = __expf(m[v] - nm);
      const float p0 = __expf(s[0][v] - nm);
      const float p1 = __expf(s[1][v] - nm);
      const float rs = red_sum16(p0 + p1);
      l[v] = l[v] * a_ + rs;
      m[v] = nm;
      alpha[v] = a_;
      Ps[w][rowoff + v][lane & 15] = (bf16_t)p0;
      Ps[w][rowoff + v][16 + (lane & 15)] = (bf16_t)p1;
    }

#pragma unroll
    for (int c = 0; c < 8; ++c)
#pragma unroll
      for (int v = 0; v < 8; ++v) o[c][v] *= alpha[v];

    wait_ds0();  // Ps cross-lane RAW

    // O += P V : preload all 8 V fragments, then 8 back-to-back WMMAs
    const v16bf pf = load_frag(&Ps[w][r0][klo], &Ps[w][r0][16 + klo]);
    v16bf vf[8];
#pragma unroll
    for (int c = 0; c < 8; ++c)
      vf[c] = load_frag(&Vt[buf][c * 16 + r0][klo],
                        &Vt[buf][c * 16 + r0][16 + klo]);
#pragma unroll
    for (int c = 0; c < 8; ++c) o[c] = wmma_bf16(pf, vf[c], o[c]);

    __syncthreads();  // buffer reuse protection
  }

  // normalize and store bf16 (A operand of the WO projection)
#pragma unroll
  for (int v = 0; v < 8; ++v) {
    const float inv = 1.0f / l[v];
#pragma unroll
    for (int c = 0; c < 8; ++c)
      out[(size_t)(q0 + rowoff + v) * (HH * HD) + h * HD + c * 16 + (lane & 15)] =
          (bf16_t)(o[c][v] * inv);
  }
}

// ---------------------------------------------------------------------------
// SwiGLU + visual-mask split, bf16 outputs:
//   at = mask ? 0 : silu(g1)*g3 ;  av = mask ? silu(g1v)*g3v : 0
// ---------------------------------------------------------------------------
__global__ __launch_bounds__(256)
void swiglu_mask_kernel(const float* __restrict__ g1, const float* __restrict__ g3,
                        const float* __restrict__ g1v, const float* __restrict__ g3v,
                        const unsigned char* __restrict__ mask,
                        bf16_t* __restrict__ at, bf16_t* __restrict__ av) {
  const size_t idx = (size_t)blockIdx.x * 256 + threadIdx.x;
  const int t = (int)(idx >> 13);  // I = 8192
  const bool vis = mask[t] != 0;
  const float a = g1[idx], b = g3[idx];
  const float a2 = g1v[idx], b2 = g3v[idx];
  const float st = a * (1.f / (1.f + __expf(-a))) * b;
  const float sv = a2 * (1.f / (1.f + __expf(-a2))) * b2;
  at[idx] = (bf16_t)(vis ? 0.f : st);
  av[idx] = (bf16_t)(vis ? sv : 0.f);
}

// ---------------------------------------------------------------------------
// Launch pipeline
// ---------------------------------------------------------------------------
extern "C" void kernel_launch(void* const* d_in, const int* in_sizes, int n_in,
                              void* d_out, int out_size, void* d_ws, size_t ws_size,
                              hipStream_t stream) {
  (void)in_sizes; (void)n_in; (void)out_size; (void)ws_size;
  const int* positions = (const int*)d_in[0];
  const float* hs = (const float*)d_in[1];
  const float* res = (const float*)d_in[2];
  const unsigned char* mask = (const unsigned char*)d_in[3];
  const float* attn_w = (const float*)d_in[4];
  const float* ffn_w = (const float*)d_in[5];
  const float* wqkv = (const float*)d_in[6];
  const float* wo = (const float*)d_in[7];
  const float* w1 = (const float*)d_in[8];
  const float* w3 = (const float*)d_in[9];
  const float* w2 = (const float*)d_in[10];
  const float* w1v = (const float*)d_in[11];
  const float* w3v = (const float*)d_in[12];
  const float* w2v = (const float*)d_in[13];

  const size_t TD = (size_t)TT * DD;        // 4M
  const size_t TI = (size_t)TT * II;        // 16M
  float* out = (float*)d_out;               // tuple elem 0: [T, D]
  float* r2 = out + TD;                     // tuple elem 1: [T, D]

  // fp32 workspace region
  float* wsf = (float*)d_ws;
  float* r1     = wsf;               // TD
  float* qkv    = wsf + TD;          // 2*TD  [T, 4096]
  float* attn_p = wsf + 3 * TD;      // TD
  float* g1     = wsf + 4 * TD;      // TI
  float* g3     = g1 + TI;
  float* g1v    = g3 + TI;
  float* g3v    = g1v + TI;

  // bf16 workspace region
  bf16_t* wsb = (bf16_t*)(g3v + TI);
  bf16_t* hb      = wsb;                  // TD
  bf16_t* attn_ob = hb + TD;              // TD
  bf16_t* h2b     = attn_ob + TD;         // TD
  bf16_t* atb     = h2b + TD;             // TI
  bf16_t* avb     = atb + TI;             // TI
  bf16_t* kpk     = avb + TI;             // K*T*HD = 2M
  bf16_t* vtp     = kpk + (size_t)KK * TT * HD;
  bf16_t* wqkv_t  = vtp + (size_t)KK * TT * HD;   // [4096, 2048]
  bf16_t* wo_t    = wqkv_t + (size_t)4096 * DD;   // [2048, 2048]
  bf16_t* w1t     = wo_t + TD;                    // [I, D]
  bf16_t* w3t     = w1t + TI;
  bf16_t* w1vt    = w3t + TI;
  bf16_t* w3vt    = w1vt + TI;
  bf16_t* w2t     = w3vt + TI;                    // [D, I]
  bf16_t* w2vt    = w2t + TI;

  // ---- weight convert + transpose (bf16) ----
  convT_kernel<<<dim3(4096 / 32, DD / 32), 256, 0, stream>>>(wqkv, wqkv_t, DD, 4096);
  convT_kernel<<<dim3(DD / 32, DD / 32), 256, 0, stream>>>(wo, wo_t, DD, DD);
  convT_kernel<<<dim3(II / 32, DD / 32), 256, 0, stream>>>(w1, w1t, DD, II);
  convT_kernel<<<dim3(II / 32, DD / 32), 256, 0, stream>>>(w3, w3t, DD, II);
  convT_kernel<<<dim3(II / 32, DD / 32), 256, 0, stream>>>(w1v, w1vt, DD, II);
  convT_kernel<<<dim3(II / 32, DD / 32), 256, 0, stream>>>(w3v, w3vt, DD, II);
  convT_kernel<<<dim3(DD / 32, II / 32), 256, 0, stream>>>(w2, w2t, II, DD);
  convT_kernel<<<dim3(DD / 32, II / 32), 256, 0, stream>>>(w2v, w2vt, II, DD);

  // ---- attention block ----
  addnorm_kernel<<<TT, 256, 0, stream>>>(hs, res, attn_w, r1, hb);
  gemm_bf16<false><<<dim3(4096 / BN, TT / BM), 256, 0, stream>>>(
      hb, wqkv_t, qkv, TT, 4096, DD);
  rope_kernel<<<TT, 256, 0, stream>>>(qkv, positions);
  packkv_kernel<<<TT, 256, 0, stream>>>(qkv, kpk, vtp);
  attn_kernel<<<dim3(TT / 64, HH), 128, 0, stream>>>(qkv, kpk, vtp, attn_ob);
  gemm_bf16<false><<<dim3(DD / BN, TT / BM), 256, 0, stream>>>(
      attn_ob, wo_t, attn_p, TT, DD, HH * HD);

  // ---- MLP block (both branches; mask-zeroed activations) ----
  addnorm_kernel<<<TT, 256, 0, stream>>>(attn_p, r1, ffn_w, r2, h2b);
  gemm_bf16<false><<<dim3(II / BN, TT / BM), 256, 0, stream>>>(h2b, w1t, g1, TT, II, DD);
  gemm_bf16<false><<<dim3(II / BN, TT / BM), 256, 0, stream>>>(h2b, w3t, g3, TT, II, DD);
  gemm_bf16<false><<<dim3(II / BN, TT / BM), 256, 0, stream>>>(h2b, w1vt, g1v, TT, II, DD);
  gemm_bf16<false><<<dim3(II / BN, TT / BM), 256, 0, stream>>>(h2b, w3vt, g3v, TT, II, DD);
  swiglu_mask_kernel<<<(unsigned)(TI / 256), 256, 0, stream>>>(
      g1, g3, g1v, g3v, mask, atb, avb);
  gemm_bf16<false><<<dim3(DD / BN, TT / BM), 256, 0, stream>>>(atb, w2t, out, TT, DD, II);
  gemm_bf16<true><<<dim3(DD / BN, TT / BM), 256, 0, stream>>>(avb, w2vt, out, TT, DD, II);
}
